// VQVAE2_29635274343091
// MI455X (gfx1250) — matrix operand order repaired and checked by reading, not compiled, not run on hardware
//
#include <hip/hip_runtime.h>

// ---------------------------------------------------------------------------
// CDNA5 (gfx1250) VQVAE-2 forward.  All convolutions (incl. transposed) are
// implicit-GEMM on v_wmma_f32_16x16x32_f16: M = Cout, N = B*Ho*Wo,
// K = Cin*KH*KW reordered TAP-MAJOR (k = (kh*KW+kw)*Cin + ci) so that when
// Cin % 32 == 0 each 32-wide K chunk sits in a single (kh,kw) tap:
// one (ih,iw,valid) computation feeds 16 channel-strided loads and MBLK wmma.
// fp32 accumulate, f16 operands.  MODE/STRIDE/MBLK are template params.
// ---------------------------------------------------------------------------

typedef __attribute__((ext_vector_type(16))) _Float16 v16h;
typedef __attribute__((ext_vector_type(8)))  float    v8f;

#define B_N 32      // batch
#define VQ_K 512    // codebook size
#define VQ_C 64     // code dim

// ---------------------------------------------------------------------------
// Weight packing: fp32 [Cout][Cin][KH][KW] -> f16 A-fragments, tap-major K.
// Packed layout: [mTile][kTile][lane(32)][16 halves]; for lane l, half i:
//   M  = mTile*16 + (l & 15)
//   k' = kTile*32 + (l>>4)*16 + i                    (A-operand layout, matches B)
//   tap = k'/Cin, ci = k'%Cin  ->  w[m][ci][tap]
// ---------------------------------------------------------------------------
__global__ __launch_bounds__(256) void pack_weights(
    const float* __restrict__ w, _Float16* __restrict__ packed,
    int Cout, int Cin, int KHKW, int Mtiles, int KT)
{
    int idx = blockIdx.x * 256 + threadIdx.x;
    int total = Mtiles * KT * 32 * 16;
    if (idx >= total) return;
    int i    = idx & 15;
    int lane = (idx >> 4) & 31;
    int rest = idx >> 9;
    int kt   = rest % KT;
    int mt   = rest / KT;
    int laneHi = lane >> 4;
    int m  = mt * 16 + (lane & 15);
    int kp = kt * 32 + laneHi * 16 + i;
    const int K = Cin * KHKW;
    float v = 0.0f;
    if (m < Cout && kp < K) {
        int tap = kp / Cin;
        int ci  = kp - tap * Cin;
        v = w[(m * Cin + ci) * KHKW + tap];
    }
    packed[idx] = (_Float16)v;
}

// ---------------------------------------------------------------------------
// Implicit-GEMM conv / transposed-conv (tap-major K).
// MODE 0: ih = oh*STRIDE - pad + kh              (regular conv)
// MODE 1: p  = oh - pad + kh ; ih = p/STRIDE, valid iff p>=0 && p%STRIDE==0
//         (lhs-dilated conv; caller passes pad = KH-1-orig_pad)
// One wave per 16-col N tile x MBLK 16-row M tiles (2 waves / block).
// B-operand (32x16 f16) per-lane layout: lane holds column n = lane&15,
// half i maps to K = (lane>>4)*16 + i within the chunk.
// ---------------------------------------------------------------------------
template <int MODE, int STRIDE, int MBLK>
__global__ __launch_bounds__(64) void conv_wmma(
    const float* __restrict__ in, const _Float16* __restrict__ Apk,
    const float* __restrict__ bias, float* __restrict__ out,
    int Cin, int H, int W, int Cout, int Ho, int Wo,
    int KH, int KW, int pad, int relu,
    int KT, int out_cstride, int out_coff)
{
    __shared__ int ktab[2048];                 // generic path only
    const int KHKW = KH * KW;
    const int K    = Cin * KHKW;
    const int HWin = H * W;
    const bool generic = (Cin & 31) != 0;      // block-uniform

    if (generic) {
        const int Kpad = KT * 32;
        for (int k = threadIdx.x; k < Kpad; k += blockDim.x) {
            int v = 0x00ffff00;                // sentinel: kh=255 -> OOB -> 0
            if (k < K) {
                int tap = k / Cin;
                int ci  = k - tap * Cin;
                int kh  = tap / KW;
                int kw  = tap - kh * KW;
                v = (ci << 16) | (kh << 8) | kw;
            }
            ktab[k] = v;
        }
        __syncthreads();
    }

    const int wave   = threadIdx.x >> 5;
    const int lane   = threadIdx.x & 31;
    const int laneHi = lane >> 4;
    const int laneLo = lane & 15;
    const int Ntot   = B_N * Ho * Wo;
    const int ntile  = blockIdx.x * 2 + wave;
    const int n0     = ntile * 16;
    if (n0 >= Ntot) return;                    // wave-uniform exit
    const int mt0 = blockIdx.y * MBLK;

    int ncol = n0 + laneLo;
    int n    = ncol < Ntot ? ncol : Ntot - 1;  // clamp; masked at store
    int b    = n / (Ho * Wo);
    int r2   = n - b * (Ho * Wo);
    int oh   = r2 / Wo;
    int ow   = r2 - oh * Wo;

    v8f acc[MBLK];
#pragma unroll
    for (int j = 0; j < MBLK; ++j) acc[j] = (v8f){0.f,0.f,0.f,0.f,0.f,0.f,0.f,0.f};

    const _Float16* aptr = Apk + ((size_t)mt0 * KT) * 512 + (size_t)lane * 16;
    const size_t aTileStride = (size_t)KT * 512;

    if (!generic) {
        // ------- fast path: each 32-chunk is one tap; (ih,iw,ok) once / chunk
        const float* inb = in + (size_t)b * Cin * HWin + (size_t)laneHi * 16 * HWin;
        int kh = 0, kw = 0, ci0 = 0;
        for (int kt = 0; kt < KT; ++kt) {
            int ih, iw; bool ok;
            if (MODE == 0) {
                ih = oh * STRIDE - pad + kh;
                iw = ow * STRIDE - pad + kw;
                ok = (ih >= 0) & (ih < H) & (iw >= 0) & (iw < W);
            } else {
                int ph = oh - pad + kh;
                int pw = ow - pad + kw;
                ih = ph / STRIDE;              // constexpr stride -> shift
                iw = pw / STRIDE;
                ok = (ph >= 0) & (pw >= 0) &
                     ((ph % STRIDE) == 0) & ((pw % STRIDE) == 0) &
                     (ih < H) & (iw < W);
            }
            const float* p = inb + (size_t)ci0 * HWin + ih * W + iw;
            union { v16h v; _Float16 e[16]; } bu;
#pragma unroll
            for (int i = 0; i < 16; ++i) {
                float v = 0.0f;
                if (ok) v = p[(size_t)i * HWin];   // consecutive channels
                bu.e[i] = (_Float16)v;
            }
#pragma unroll
            for (int j = 0; j < MBLK; ++j) {
                v16h a = *(const v16h*)(aptr + j * aTileStride);
                acc[j] = __builtin_amdgcn_wmma_f32_16x16x32_f16(
                    false, a, false, bu.v, (short)0, acc[j], false, false);
            }
            aptr += 512;
            ci0 += 32;
            if (ci0 >= Cin) {                  // advance tap (incremental)
                ci0 -= Cin;
                if (++kw == KW) { kw = 0; ++kh; }
            }
        }
    } else {
        // ------- generic path (Cin % 32 != 0; only the 1-channel stem)
        const int inB = b * Cin;
        for (int kt = 0; kt < KT; ++kt) {
            union { v16h v; _Float16 e[16]; } bu;
            const int kbase = kt * 32 + laneHi * 16;
#pragma unroll
            for (int i = 0; i < 16; ++i) {
                int t  = ktab[kbase + i];
                int ci = t >> 16, kh = (t >> 8) & 255, kw = t & 255;
                int ih, iw; bool ok;
                if (MODE == 0) {
                    ih = oh * STRIDE - pad + kh;
                    iw = ow * STRIDE - pad + kw;
                    ok = (ih >= 0) & (ih < H) & (iw >= 0) & (iw < W);
                } else {
                    int ph = oh - pad + kh;
                    int pw = ow - pad + kw;
                    ih = ph / STRIDE;
                    iw = pw / STRIDE;
                    ok = (ph >= 0) & (pw >= 0) &
                         ((ph % STRIDE) == 0) & ((pw % STRIDE) == 0) &
                         (ih < H) & (iw < W);
                }
                float v = 0.0f;
                if (ok) v = in[((inB + ci) * H + ih) * W + iw];
                bu.e[i] = (_Float16)v;
            }
#pragma unroll
            for (int j = 0; j < MBLK; ++j) {
                v16h a = *(const v16h*)(aptr + j * aTileStride);
                acc[j] = __builtin_amdgcn_wmma_f32_16x16x32_f16(
                    false, a, false, bu.v, (short)0, acc[j], false, false);
            }
            aptr += 512;
        }
    }

    if (ncol < Ntot) {
#pragma unroll
        for (int j = 0; j < MBLK; ++j) {
#pragma unroll
            for (int rI = 0; rI < 8; ++rI) {
                int m = (mt0 + j) * 16 + laneHi * 8 + rI;  // C/D: M = r + 8*laneHi
                if (m < Cout) {
                    float v = acc[j][rI];
                    if (bias) v += bias[m];
                    if (relu) v = fmaxf(v, 0.0f);
                    out[((b * out_cstride + out_coff + m) * Ho + oh) * Wo + ow] = v;
                }
            }
        }
    }
}

// ---------------------------------------------------------------------------
// VQ: per-vector argmin over 512 codes (C=64), first-min tie break.
// ---------------------------------------------------------------------------
__global__ __launch_bounds__(64) void vq_argmin(
    const float* __restrict__ z, const float* __restrict__ emb,
    int* __restrict__ idx, int* __restrict__ hist, int HW)
{
    const int v = blockIdx.x;
    __shared__ float zs[VQ_C];
    __shared__ float bd[64];
    __shared__ int   bi[64];
    const int t = threadIdx.x;
    const int b = v / HW;
    const int pos = v - b * HW;
    zs[t] = z[(b * VQ_C + t) * HW + pos];
    __syncthreads();

    float best = 3.4e38f; int bk = 0;
    for (int k = t; k < VQ_K; k += 64) {
        const float* e = emb + k * VQ_C;
        float d = 0.0f;
#pragma unroll
        for (int c = 0; c < VQ_C; ++c) {
            float df = zs[c] - e[c];
            d = fmaf(df, df, d);
        }
        if (d < best) { best = d; bk = k; }
    }
    bd[t] = best; bi[t] = bk;
    __syncthreads();
    for (int s = 32; s > 0; s >>= 1) {
        if (t < s) {
            if (bd[t + s] < bd[t] || (bd[t + s] == bd[t] && bi[t + s] < bi[t])) {
                bd[t] = bd[t + s]; bi[t] = bi[t + s];
            }
        }
        __syncthreads();
    }
    if (t == 0) {
        idx[v] = bi[0];
        atomicAdd(&hist[bi[0]], 1);
    }
}

// q = emb[idx] written into (possibly concat) dst; accumulates 1.25*mean((q-z)^2)
__global__ __launch_bounds__(256) void vq_gather(
    const float* __restrict__ z, const float* __restrict__ emb,
    const int* __restrict__ idx, float* __restrict__ q,
    float* __restrict__ loss_out, int HW, int dstC, int dstOff, float scale)
{
    const int total = B_N * HW * VQ_C;
    const int i = blockIdx.x * 256 + threadIdx.x;
    float sq = 0.0f;
    if (i < total) {
        int c = i % VQ_C;
        int v = i / VQ_C;
        int b = v / HW;
        int pos = v - b * HW;
        float zv = z[(b * VQ_C + c) * HW + pos];
        float qv = emb[idx[v] * VQ_C + c];
        q[(b * dstC + dstOff + c) * HW + pos] = qv;
        float d = qv - zv;
        sq = d * d;
    }
    __shared__ float red[256];
    red[threadIdx.x] = sq;
    __syncthreads();
    for (int s = 128; s > 0; s >>= 1) {
        if (threadIdx.x < s) red[threadIdx.x] += red[threadIdx.x + s];
        __syncthreads();
    }
    if (threadIdx.x == 0) atomicAdd(loss_out, red[0] * scale);
}

__global__ __launch_bounds__(512) void vq_perplexity(
    const int* __restrict__ hist, float* __restrict__ out, float invN)
{
    __shared__ float red[VQ_K];
    const int t = threadIdx.x;
    float p = (float)hist[t] * invN;
    red[t] = p * logf(p + 1e-10f);
    __syncthreads();
    for (int s = 256; s > 0; s >>= 1) {
        if (t < s) red[t] += red[t + s];
        __syncthreads();
    }
    if (t == 0) atomicAdd(out, expf(-red[0]));
}

__global__ __launch_bounds__(256) void init_scalars(
    float* loss, float* perp, int* hist, int n)
{
    int i = blockIdx.x * 256 + threadIdx.x;
    if (i == 0) { *loss = 0.0f; *perp = 0.0f; }
    if (i < n) hist[i] = 0;
}

// channel-concat copy: src [B,C,HW] -> dst [B,dstC,HW] at channel offset
__global__ __launch_bounds__(256) void copy_channels(
    const float* __restrict__ src, float* __restrict__ dst,
    int total, int C, int HW, int dstC, int dstOff)
{
    int i = blockIdx.x * 256 + threadIdx.x;
    if (i >= total) return;
    int pos = i % HW;
    int r   = i / HW;
    int c   = r % C;
    int b   = r / C;
    dst[(b * dstC + dstOff + c) * HW + pos] = src[i];
}

// ---------------------------------------------------------------------------
// Host-side helper: pack weights then dispatch the right specialization.
// ---------------------------------------------------------------------------
static void run_conv(hipStream_t stream,
                     const float* in, const float* w, const float* bias,
                     float* out, _Float16* wpack,
                     int Cin, int H, int W, int Cout, int KH, int KW,
                     int stride, int pad, int mode, int relu,
                     int Ho, int Wo, int out_cstride, int out_coff)
{
    const int K  = Cin * KH * KW;
    const int KT = (K + 31) / 32;
    const int Mtiles = (Cout + 15) / 16;
    const int packTotal = Mtiles * KT * 512;
    pack_weights<<<(packTotal + 255) / 256, 256, 0, stream>>>(
        w, wpack, Cout, Cin, KH * KW, Mtiles, KT);

    const int Ntot = B_N * Ho * Wo;
    const int ntiles = (Ntot + 15) / 16;
    const int mblk = (Mtiles % 4 == 0) ? 4 : ((Mtiles % 2 == 0) ? 2 : 1);
    dim3 grid((ntiles + 1) / 2, Mtiles / mblk);

#define ARGS in, wpack, bias, out, Cin, H, W, Cout, Ho, Wo, KH, KW, pad, relu, KT, out_cstride, out_coff
    if (mode == 0 && stride == 1) {
        if      (mblk == 4) conv_wmma<0,1,4><<<grid, 64, 0, stream>>>(ARGS);
        else if (mblk == 2) conv_wmma<0,1,2><<<grid, 64, 0, stream>>>(ARGS);
        else                conv_wmma<0,1,1><<<grid, 64, 0, stream>>>(ARGS);
    } else if (mode == 0) {  // stride 2
        if      (mblk == 4) conv_wmma<0,2,4><<<grid, 64, 0, stream>>>(ARGS);
        else if (mblk == 2) conv_wmma<0,2,2><<<grid, 64, 0, stream>>>(ARGS);
        else                conv_wmma<0,2,1><<<grid, 64, 0, stream>>>(ARGS);
    } else {                 // transposed, stride 2
        if      (mblk == 4) conv_wmma<1,2,4><<<grid, 64, 0, stream>>>(ARGS);
        else if (mblk == 2) conv_wmma<1,2,2><<<grid, 64, 0, stream>>>(ARGS);
        else                conv_wmma<1,2,1><<<grid, 64, 0, stream>>>(ARGS);
    }
#undef ARGS
}

// ---------------------------------------------------------------------------
extern "C" void kernel_launch(void* const* d_in, const int* in_sizes, int n_in,
                              void* d_out, int out_size, void* d_ws, size_t ws_size,
                              hipStream_t stream)
{
    (void)in_sizes; (void)n_in; (void)out_size; (void)ws_size;

    // --- inputs (setup_inputs dict order: x, then params in insertion order)
    const float* x       = (const float*)d_in[0];
    const float* eb_w1   = (const float*)d_in[1];
    const float* eb_b1   = (const float*)d_in[2];
    const float* eb_w2   = (const float*)d_in[3];
    const float* eb_b2   = (const float*)d_in[4];
    const float* eb_w3   = (const float*)d_in[5];
    const float* eb_b3   = (const float*)d_in[6];
    const float* eb_r1w1 = (const float*)d_in[7];
    const float* eb_r1w2 = (const float*)d_in[8];
    const float* eb_r2w1 = (const float*)d_in[9];
    const float* eb_r2w2 = (const float*)d_in[10];
    const float* et_w1   = (const float*)d_in[11];
    const float* et_b1   = (const float*)d_in[12];
    const float* et_w2   = (const float*)d_in[13];
    const float* et_b2   = (const float*)d_in[14];
    const float* et_r1w1 = (const float*)d_in[15];
    const float* et_r1w2 = (const float*)d_in[16];
    const float* et_r2w1 = (const float*)d_in[17];
    const float* et_r2w2 = (const float*)d_in[18];
    const float* pvt_w   = (const float*)d_in[19];
    const float* pvt_b   = (const float*)d_in[20];
    const float* pvb_w   = (const float*)d_in[21];
    const float* pvb_b   = (const float*)d_in[22];
    const float* emb_top = (const float*)d_in[23];
    const float* emb_bot = (const float*)d_in[24];
    const float* dt_w1   = (const float*)d_in[25];
    const float* dt_b1   = (const float*)d_in[26];
    const float* dt_r1w1 = (const float*)d_in[27];
    const float* dt_r1w2 = (const float*)d_in[28];
    const float* dt_r2w1 = (const float*)d_in[29];
    const float* dt_r2w2 = (const float*)d_in[30];
    const float* dt_tw   = (const float*)d_in[31];
    const float* dt_tb   = (const float*)d_in[32];
    const float* up_w    = (const float*)d_in[33];
    const float* up_b    = (const float*)d_in[34];
    const float* db_w1   = (const float*)d_in[35];
    const float* db_b1   = (const float*)d_in[36];
    const float* db_r1w1 = (const float*)d_in[37];
    const float* db_r1w2 = (const float*)d_in[38];
    const float* db_r2w1 = (const float*)d_in[39];
    const float* db_r2w2 = (const float*)d_in[40];
    const float* db_tw1  = (const float*)d_in[41];
    const float* db_tb1  = (const float*)d_in[42];
    const float* db_tw2  = (const float*)d_in[43];
    const float* db_tb2  = (const float*)d_in[44];

    // --- output layout: loss | x_rec[32*1*252*252] | perp | quant[32*128*63*63]
    float* out_loss  = (float*)d_out;
    float* out_xrec  = out_loss + 1;
    float* out_perp  = out_loss + 1 + 32 * 252 * 252;
    float* out_quant = out_perp + 1;

    // --- workspace bump allocator
    char* wsBase = (char*)d_ws;
    size_t off = 0;
    auto alloc = [&](size_t bytes) -> void* {
        void* p = wsBase + off;
        off = (off + bytes + 255) & ~(size_t)255;
        return p;
    };
    const size_t S63_128 = (size_t)32 * 128 * 63 * 63;
    const size_t S63_32  = (size_t)32 * 32 * 63 * 63;
    const size_t S32_128 = (size_t)32 * 128 * 32 * 32;
    const size_t S32_64  = (size_t)32 * 64 * 32 * 32;
    const size_t S32_32  = (size_t)32 * 32 * 32 * 32;

    float* P      = (float*)alloc(sizeof(float) * 32ull * 64 * 126 * 126);
    float* Q      = (float*)alloc(sizeof(float) * S63_128);
    float* R      = (float*)alloc(sizeof(float) * S63_128);
    float* S      = (float*)alloc(sizeof(float) * S63_128);
    float* Zb     = (float*)alloc(sizeof(float) * S63_128);
    float* C32a   = (float*)alloc(sizeof(float) * S63_32);
    float* C32b   = (float*)alloc(sizeof(float) * S63_32);
    float* U64a   = (float*)alloc(sizeof(float) * S32_64);
    float* q_top  = (float*)alloc(sizeof(float) * S32_64);
    float* U128a  = (float*)alloc(sizeof(float) * S32_128);
    float* U128b  = (float*)alloc(sizeof(float) * S32_128);
    float* U128c  = (float*)alloc(sizeof(float) * S32_128);
    float* U32a   = (float*)alloc(sizeof(float) * S32_32);
    float* zbcat  = (float*)alloc(sizeof(float) * 32ull * 192 * 63 * 63);
    float* zb_in  = (float*)alloc(sizeof(float) * 32ull * 64 * 63 * 63);
    int*   idx_t  = (int*)alloc(sizeof(int) * 32ull * 32 * 32);
    int*   idx_b  = (int*)alloc(sizeof(int) * 32ull * 63 * 63);
    int*   hist_t = (int*)alloc(sizeof(int) * VQ_K);
    int*   hist_b = (int*)alloc(sizeof(int) * VQ_K);
    _Float16* wpack = (_Float16*)alloc(sizeof(_Float16) * 262144);

    // zero loss/perp/histograms (d_out is poisoned between runs)
    init_scalars<<<4, 256, 0, stream>>>(out_loss, out_perp, hist_t, VQ_K);
    init_scalars<<<4, 256, 0, stream>>>(out_loss, out_perp, hist_b, VQ_K);

    // ---- encoder_bottom
    run_conv(stream, x,    eb_w1, eb_b1, P, wpack,   1, 252, 252, 64, 4, 4, 2, 1, 0, 1, 126, 126, 64, 0);
    run_conv(stream, P,    eb_w2, eb_b2, Q, wpack,  64, 126, 126, 128, 4, 4, 2, 1, 0, 1, 63, 63, 128, 0);
    run_conv(stream, Q,    eb_w3, eb_b3, R, wpack, 128, 63, 63, 128, 3, 3, 1, 1, 0, 1, 63, 63, 128, 0);
    run_conv(stream, R,    eb_r1w1, nullptr, C32a, wpack, 128, 63, 63, 32, 3, 3, 1, 1, 0, 1, 63, 63, 32, 0);
    run_conv(stream, C32a, eb_r1w2, nullptr, Q,    wpack,  32, 63, 63, 128, 1, 1, 1, 0, 0, 1, 63, 63, 128, 0);
    run_conv(stream, Q,    eb_r2w1, nullptr, C32b, wpack, 128, 63, 63, 32, 3, 3, 1, 1, 0, 1, 63, 63, 32, 0);
    run_conv(stream, C32b, eb_r2w2, nullptr, Zb,   wpack,  32, 63, 63, 128, 1, 1, 1, 0, 0, 1, 63, 63, 128, 0);

    // ---- encoder_top
    run_conv(stream, Zb,    et_w1, et_b1, U64a,  wpack, 128, 63, 63, 64, 3, 3, 2, 1, 0, 1, 32, 32, 64, 0);
    run_conv(stream, U64a,  et_w2, et_b2, U128a, wpack,  64, 32, 32, 128, 3, 3, 1, 1, 0, 1, 32, 32, 128, 0);
    run_conv(stream, U128a, et_r1w1, nullptr, U32a,  wpack, 128, 32, 32, 32, 3, 3, 1, 1, 0, 1, 32, 32, 32, 0);
    run_conv(stream, U32a,  et_r1w2, nullptr, U128b, wpack,  32, 32, 32, 128, 1, 1, 1, 0, 0, 1, 32, 32, 128, 0);
    run_conv(stream, U128b, et_r2w1, nullptr, U32a,  wpack, 128, 32, 32, 32, 3, 3, 1, 1, 0, 1, 32, 32, 32, 0);
    run_conv(stream, U32a,  et_r2w2, nullptr, U128c, wpack,  32, 32, 32, 128, 1, 1, 1, 0, 0, 1, 32, 32, 128, 0);

    // ---- top VQ: pre_vq conv, argmin, gather (q_top), loss
    run_conv(stream, U128c, pvt_w, pvt_b, U64a, wpack, 128, 32, 32, 64, 1, 1, 1, 0, 0, 0, 32, 32, 64, 0);
    {
        const int HW = 32 * 32, Nvec = B_N * HW;
        vq_argmin<<<Nvec, 64, 0, stream>>>(U64a, emb_top, idx_t, hist_t, HW);
        const int tot = Nvec * VQ_C;
        vq_gather<<<(tot + 255) / 256, 256, 0, stream>>>(
            U64a, emb_top, idx_t, q_top, out_loss, HW, 64, 0, 1.25f / (float)tot);
    }

    // ---- decoder_top (no activation on plain convs)
    run_conv(stream, q_top, dt_w1, dt_b1, U128a, wpack,  64, 32, 32, 128, 3, 3, 1, 1, 0, 0, 32, 32, 128, 0);
    run_conv(stream, U128a, dt_r1w1, nullptr, U32a,  wpack, 128, 32, 32, 32, 3, 3, 1, 1, 0, 1, 32, 32, 32, 0);
    run_conv(stream, U32a,  dt_r1w2, nullptr, U128b, wpack,  32, 32, 32, 128, 1, 1, 1, 0, 0, 1, 32, 32, 128, 0);
    run_conv(stream, U128b, dt_r2w1, nullptr, U32a,  wpack, 128, 32, 32, 32, 3, 3, 1, 1, 0, 1, 32, 32, 32, 0);
    run_conv(stream, U32a,  dt_r2w2, nullptr, U128c, wpack,  32, 32, 32, 128, 1, 1, 1, 0, 0, 1, 32, 32, 128, 0);
    // rec_top = convT(d, dt_tw, s2, k3, pad1 -> q=1) -> zbcat[:, 0:64]
    run_conv(stream, U128c, dt_tw, dt_tb, zbcat, wpack, 128, 32, 32, 64, 3, 3, 2, 1, 1, 0, 63, 63, 192, 0);
    // zbcat[:, 64:192] = z_bottom
    {
        const int tot = (int)S63_128;
        copy_channels<<<(tot + 255) / 256, 256, 0, stream>>>(
            Zb, zbcat, tot, 128, 63 * 63, 192, 64);
    }

    // ---- bottom VQ
    run_conv(stream, zbcat, pvb_w, pvb_b, zb_in, wpack, 192, 63, 63, 64, 1, 1, 1, 0, 0, 0, 63, 63, 64, 0);
    {
        const int HW = 63 * 63, Nvec = B_N * HW;
        vq_argmin<<<Nvec, 64, 0, stream>>>(zb_in, emb_bot, idx_b, hist_b, HW);
        const int tot = Nvec * VQ_C;
        // q_bot -> quant[:, 64:128] (directly in d_out)
        vq_gather<<<(tot + 255) / 256, 256, 0, stream>>>(
            zb_in, emb_bot, idx_b, out_quant, out_loss, HW, 128, 64, 1.25f / (float)tot);
    }
    // up_top = convT(q_top, up_w, s2, k3, pad1 -> q=1) -> quant[:, 0:64]
    run_conv(stream, q_top, up_w, up_b, out_quant, wpack, 64, 32, 32, 64, 3, 3, 2, 1, 1, 0, 63, 63, 128, 0);

    // perplexity (top + bottom summed into one scalar)
    vq_perplexity<<<1, VQ_K, 0, stream>>>(hist_t, out_perp, 1.0f / (32.0f * 32 * 32));
    vq_perplexity<<<1, VQ_K, 0, stream>>>(hist_b, out_perp, 1.0f / (32.0f * 63 * 63));

    // ---- decoder_bottom (reads quant straight from d_out)
    run_conv(stream, out_quant, db_w1, db_b1, Q, wpack, 128, 63, 63, 128, 3, 3, 1, 1, 0, 0, 63, 63, 128, 0);
    run_conv(stream, Q,    db_r1w1, nullptr, C32a, wpack, 128, 63, 63, 32, 3, 3, 1, 1, 0, 1, 63, 63, 32, 0);
    run_conv(stream, C32a, db_r1w2, nullptr, R,    wpack,  32, 63, 63, 128, 1, 1, 1, 0, 0, 1, 63, 63, 128, 0);
    run_conv(stream, R,    db_r2w1, nullptr, C32a, wpack, 128, 63, 63, 32, 3, 3, 1, 1, 0, 1, 63, 63, 32, 0);
    run_conv(stream, C32a, db_r2w2, nullptr, S,    wpack,  32, 63, 63, 128, 1, 1, 1, 0, 0, 1, 63, 63, 128, 0);
    // relu(convT(., db_tw1, s2, k4, pad1 -> q=2)) -> [32,64,126,126]
    run_conv(stream, S, db_tw1, db_tb1, P, wpack, 128, 63, 63, 64, 4, 4, 2, 2, 1, 1, 126, 126, 64, 0);
    // x_rec = convT(., db_tw2, s2, k4, pad1 -> q=2) -> [32,1,252,252] (into d_out)
    run_conv(stream, P, db_tw2, db_tb2, out_xrec, wpack, 64, 126, 126, 1, 4, 4, 2, 2, 1, 0, 252, 252, 1, 0);
}
